// EncoderModelLSH_40578851013251
// MI455X (gfx1250) — compile-verified, matrix operand order checked by reading, not run on hardware
//
#include <hip/hip_runtime.h>
#include <hip/hip_bf16.h>
#include <math.h>

// ---------------------------------------------------------------------------
// MI455X (gfx1250, wave32) LSH-encoder.
// Q@Q^T is bucket-independent (mask only zeroes entries; masked logits are 0
// INSIDE the softmax): logits computed once per encoder; key-mask folded into
// the normalized P, making V bucket-independent too.
// GEMMs: v_wmma_f32_16x16x32_bf16 (f32 accum).  Weights are pre-transposed to
// [N][K] at bf16-conversion time so GEMM staging is one fully-vectorized path.
// Attention logit stripe is fetched by the Tensor Data Mover
// (tensor_load_to_lds + s_wait_tensorcnt).
// ---------------------------------------------------------------------------

typedef __attribute__((ext_vector_type(16))) __bf16   v16bf;
typedef __attribute__((ext_vector_type(8)))  __bf16   v8bf;
typedef __attribute__((ext_vector_type(8)))  float    v8f;
typedef __attribute__((ext_vector_type(4)))  unsigned v4u;
typedef __attribute__((ext_vector_type(8)))  int      v8i;
typedef __attribute__((ext_vector_type(4)))  int      v4i;

#define D_    128
#define S_    512
#define B_    8
#define HID_  512
#define NBUCK_ 32
#define NHYP_  5

// bf16 fragment: lanes 0-15 hold K={0..7,16..23}, lanes 16-31 K={8..15,24..31}
// => per lane two contiguous 16-byte runs of a K-contiguous row.
__device__ __forceinline__ v16bf ldfrag(const __bf16* base, int hf) {
  union { v16bf v; v8bf h[2]; } u;
  u.h[0] = *(const v8bf*)(base + (hf << 3));
  u.h[1] = *(const v8bf*)(base + 16 + (hf << 3));
  return u.v;
}

// ---------------------------------------------------------------------------
// bf16 WMMA GEMM: C[M,N] = alpha*A[M,K]@B + bias/relu/resid.
// A row-major [M][K] (lda);  Bt is the K x N operand stored TRANSPOSED as
// [N][K] rows (ldb).  Batched via grid.z.  WG 128 thr / 4 waves, 64x64 tile,
// each wave a 32x32 quadrant.  LDS: sA[m][k], sB[n][k]; every fragment is two
// ds_load_b128.
// ---------------------------------------------------------------------------
__global__ __launch_bounds__(128)
void gemm_bf16(const __bf16* __restrict__ A, const __bf16* __restrict__ Bt,
               const float* __restrict__ bias, const float* __restrict__ resid,
               float* __restrict__ Cf, __bf16* __restrict__ Cb,
               int M, int N, int K, int lda, int ldb, int ldc,
               long sA_, long sB_, long sC_, int doRelu, float alpha)
{
  __shared__ __align__(16) __bf16 sA[64][40];   // 80B pitch, 16B-aligned rows
  __shared__ __align__(16) __bf16 sB[64][40];

  const int tid  = threadIdx.x;
  const int lane = tid & 31;
  const int w    = tid >> 5;
  const int hf   = lane >> 4;
  const int l15  = lane & 15;
  const int wm   = w >> 1, wn = w & 1;

  const int mBase = blockIdx.y * 64;
  const int nBase = blockIdx.x * 64;
  const int batch = blockIdx.z;
  A  += (long)batch * sA_;
  Bt += (long)batch * sB_;
  if (Cf)    Cf    += (long)batch * sC_;
  if (Cb)    Cb    += (long)batch * sC_;
  if (resid) resid += (long)batch * sC_;

  v8f acc[2][2];
  #pragma unroll
  for (int i = 0; i < 2; ++i)
    #pragma unroll
    for (int j = 0; j < 2; ++j)
      acc[i][j] = (v8f){0.f,0.f,0.f,0.f,0.f,0.f,0.f,0.f};

  for (int kb = 0; kb < K; kb += 32) {
    // stage A 64x32 and B(t) 64x32: 2 x 16B chunks per thread each, all vector
    #pragma unroll
    for (int i = 0; i < 2; ++i) {
      int chunk = tid * 2 + i;
      int rr = chunk >> 2, cg = chunk & 3;
      *(v8bf*)&sA[rr][cg * 8] =
          *(const v8bf*)(A + (long)(mBase + rr) * lda + kb + cg * 8);
      *(v8bf*)&sB[rr][cg * 8] =
          *(const v8bf*)(Bt + (long)(nBase + rr) * ldb + kb + cg * 8);
    }
    __syncthreads();

    if (kb + 32 < K)     // global_prefetch_b8 next A tile
      __builtin_prefetch(&A[(long)(mBase + (tid >> 1)) * lda + kb + 32], 0, 1);

    v16bf af[2], bfr[2];
    #pragma unroll
    for (int ti = 0; ti < 2; ++ti)
      af[ti] = ldfrag(&sA[wm * 32 + ti * 16 + l15][0], hf);
    #pragma unroll
    for (int tj = 0; tj < 2; ++tj)
      bfr[tj] = ldfrag(&sB[wn * 32 + tj * 16 + l15][0], hf);

    #pragma unroll
    for (int ti = 0; ti < 2; ++ti)
      #pragma unroll
      for (int tj = 0; tj < 2; ++tj)
        acc[ti][tj] = __builtin_amdgcn_wmma_f32_16x16x32_bf16(
            false, af[ti], false, bfr[tj], (short)0, acc[ti][tj], false, false);
    __syncthreads();
  }

  #pragma unroll
  for (int ti = 0; ti < 2; ++ti)
    #pragma unroll
    for (int tj = 0; tj < 2; ++tj)
      #pragma unroll
      for (int g = 0; g < 8; ++g) {
        int row = mBase + wm * 32 + ti * 16 + g + 8 * hf;   // C/D layout
        int col = nBase + wn * 32 + tj * 16 + l15;
        float v = acc[ti][tj][g] * alpha;
        if (bias)   v += bias[col];
        if (doRelu) v = fmaxf(v, 0.f);
        if (resid)  v += resid[(long)row * ldc + col];
        if (Cf) Cf[(long)row * ldc + col] = v;
        if (Cb) Cb[(long)row * ldc + col] = (__bf16)v;
      }
}

// ---------------------------------------------------------------------------
// Fused 32-bucket masked softmax + P@V.
// grid = (S/16, B), WG = 128 thr (4 waves), 16 query rows per WG.
// Dynamic LDS (185 KB; CDNA5 allows 320 KB/WG); no static __shared__, so the
// dynamic block starts at LDS offset 0 (used by the TDM descriptor).
//   sLg [16][512] f32 : logit stripe, fetched by TENSOR_LOAD_TO_LDS
//   sP  [16][520] bf16: normalized, key-masked P (keep[t] folded in)
//   sVT [128][520]bf16: V transposed [d][token], staged ONCE (bucket-indep.)
// PV inner loop = 6x ds_load_b128 + 2x v_wmma, no barriers.
// ---------------------------------------------------------------------------
#define LDS_LG   0
#define LDS_KEEP (16*512*4)                    // 32768
#define LDS_RED  (LDS_KEEP + 512*4)            // 34816
#define LDS_P    (LDS_RED + 16*8*4)            // 35328 (16B aligned)
#define LDS_VT   (LDS_P + 16*520*2)            // 51968
#define LDS_TOT  (LDS_VT + 128*520*2)          // 185088
#define PP 520

__global__ __launch_bounds__(128)
void lsh_attn(const float* __restrict__ Lg, const __bf16* __restrict__ Vb,
              const int* __restrict__ buckets, float* __restrict__ Out)
{
  extern __shared__ __align__(16) char smem[];
  float*  sLg  = (float*) (smem + LDS_LG);
  float*  sKeep= (float*) (smem + LDS_KEEP);
  float*  sRed = (float*) (smem + LDS_RED);
  __bf16* sP   = (__bf16*)(smem + LDS_P);
  __bf16* sVT  = (__bf16*)(smem + LDS_VT);

  const int tid  = threadIdx.x;
  const int lane = tid & 31, w = tid >> 5;
  const int hf   = lane >> 4, l15 = lane & 15;
  const int qBase = blockIdx.x * 16;
  const int b     = blockIdx.y;
  const long tokBase = (long)b * S_;

  // --- TDM: DMA the 16x512 f32 logit stripe into LDS (wave 0 issues) -------
  if (tid < 32) {
    unsigned long long ga = (unsigned long long)(Lg + (tokBase + qBase) * S_);
    v4u g0;                                    // D# group 0 (ISA 8.3)
    g0.x = 1u;                                 // count=1, user descriptor
    g0.y = (unsigned)LDS_LG;                   // lds_addr (dynamic LDS base=0)
    g0.z = (unsigned)ga;                       // global_addr[31:0]
    g0.w = (unsigned)(ga >> 32) | (2u << 30);  // global_addr[56:32] | type=2
    v8i g1;                                    // D# group 1 (ISA 8.4)
    g1[0] = (int)(2u << 16);                   // wg_mask=0, data_size=4B (2)
    g1[1] = (int)(512u << 16);                 // tensor_dim0 = 512 (lo16 @63:48)
    g1[2] = (int)(16u << 16);                  // tensor_dim0 hi=0, tensor_dim1=16
    g1[3] = (int)(512u << 16);                 // tensor_dim1 hi=0, tile_dim0=512
    g1[4] = 16;                                // tile_dim1=16, tile_dim2=0
    g1[5] = 512;                               // tensor_dim0_stride lo32 = 512
    g1[6] = 0;                                 // stride hi, dim1_stride lo
    g1[7] = 0;
    v4i zz4 = {0, 0, 0, 0};                    // groups 2/3 unused (2D tensor)
    v8i zz8 = {0, 0, 0, 0, 0, 0, 0, 0};        // 6-arg form extra group
    __builtin_amdgcn_tensor_load_to_lds(g0, g1, zz4, zz4, zz8, 0);
  }

  // --- V transposed [d][token], staged once (V is bucket-independent) ------
  for (int c = 0; c < 64; ++c) {
    int chunk = tid * 64 + c;                  // 0..8191
    int tokn = chunk >> 4, dg = chunk & 15;
    v8bf vv = *(const v8bf*)(Vb + (tokBase + tokn) * D_ + dg * 8);
    #pragma unroll
    for (int j = 0; j < 8; ++j) sVT[(dg * 8 + j) * PP + tokn] = vv[j];
  }
  if (tid < 32) __builtin_amdgcn_s_wait_tensorcnt(0);   // TDM done before barrier

  v8f acc[2];
  acc[0] = (v8f){0.f,0.f,0.f,0.f,0.f,0.f,0.f,0.f};
  acc[1] = (v8f){0.f,0.f,0.f,0.f,0.f,0.f,0.f,0.f};

  const int r  = tid >> 3;                     // softmax: 8 threads per row
  const int lc = tid & 7;
  const int cb = lc * 64;

  for (int bkt = 0; bkt < NBUCK_; ++bkt) {
    __syncthreads();                           // staging done / prev PV reads done
    for (int i = tid; i < S_; i += 128)
      sKeep[i] = (buckets[tokBase + i] != bkt) ? 1.f : 0.f;
    __syncthreads();

    // masked logit = keep[s]*keep[t]*Lg  (masked entries are 0, IN softmax)
    const float kq = sKeep[qBase + r];
    float mloc = -1e30f;
    for (int j = 0; j < 64; ++j)
      mloc = fmaxf(mloc, kq * sKeep[cb + j] * sLg[r * 512 + cb + j]);
    sRed[r * 8 + lc] = mloc;
    __syncthreads();
    float rm = sRed[r * 8];
    #pragma unroll
    for (int i = 1; i < 8; ++i) rm = fmaxf(rm, sRed[r * 8 + i]);
    __syncthreads();
    float ssum = 0.f;
    for (int j = 0; j < 64; ++j)
      ssum += __expf(kq * sKeep[cb + j] * sLg[r * 512 + cb + j] - rm);
    sRed[r * 8 + lc] = ssum;
    __syncthreads();
    float tot = 0.f;
    #pragma unroll
    for (int i = 0; i < 8; ++i) tot += sRed[r * 8 + i];
    const float inv = 1.f / tot;
    // write P normalized AND key-masked: p' = keep[t]*exp(l-rm)/sum
    for (int j = 0; j < 64; ++j) {
      float v = kq * sKeep[cb + j] * sLg[r * 512 + cb + j];
      sP[r * PP + cb + j] = (__bf16)(sKeep[cb + j] * __expf(v - rm) * inv);
    }
    __syncthreads();

    // P@V: pure fragment loads + WMMA, accumulate across buckets
    for (int kc = 0; kc < 16; ++kc) {
      const int kB = kc * 32;
      v16bf a  = ldfrag(sP  + l15 * PP + kB, hf);
      v16bf b0 = ldfrag(sVT + (w * 32 + l15) * PP + kB, hf);
      v16bf b1 = ldfrag(sVT + (w * 32 + 16 + l15) * PP + kB, hf);
      acc[0] = __builtin_amdgcn_wmma_f32_16x16x32_bf16(false, a, false, b0, (short)0, acc[0], false, false);
      acc[1] = __builtin_amdgcn_wmma_f32_16x16x32_bf16(false, a, false, b1, (short)0, acc[1], false, false);
    }
  }

  #pragma unroll
  for (int tj = 0; tj < 2; ++tj)
    #pragma unroll
    for (int g = 0; g < 8; ++g) {
      int row = qBase + g + 8 * hf;
      int col = w * 32 + tj * 16 + l15;
      Out[(tokBase + row) * D_ + col] = acc[tj][g];
    }
}

// ---------------------------------------------------------------------------
// Small support kernels
// ---------------------------------------------------------------------------
__global__ __launch_bounds__(128)
void embed_k(const int* __restrict__ tok, const float* __restrict__ emb,
             const float* __restrict__ pe, float* __restrict__ xo,
             __bf16* __restrict__ xbo)
{
  const int t = blockIdx.x, d = threadIdx.x;
  const int b = t >> 9;                  // pe indexed by batch (faithful quirk)
  const long id = tok[t];
  float v = emb[id * D_ + d] + pe[b * D_ + d];
  xo[(long)t * D_ + d]  = v;
  xbo[(long)t * D_ + d] = (__bf16)v;
}

__global__ __launch_bounds__(128)
void bucket_k(const float* __restrict__ q, const float* __restrict__ hyp,
              int* __restrict__ bk)
{
  int t = blockIdx.x * 128 + threadIdx.x;      // 4096 tokens
  float acc[NHYP_];
  #pragma unroll
  for (int n = 0; n < NHYP_; ++n) acc[n] = hyp[D_ * NHYP_ + n];  // homogeneous 1
  for (int d = 0; d < D_; ++d) {
    float qv = q[(long)t * D_ + d];
    #pragma unroll
    for (int n = 0; n < NHYP_; ++n) acc[n] += qv * hyp[d * NHYP_ + n];
  }
  int id = 0;
  #pragma unroll
  for (int n = 0; n < NHYP_; ++n) if (acc[n] >= 0.f) id |= (1 << n);
  bk[t] = id;
}

__global__ __launch_bounds__(128)
void layernorm_k(const float* __restrict__ in, const float* __restrict__ g,
                 const float* __restrict__ bta, float* __restrict__ xo,
                 __bf16* __restrict__ xbo)
{
  __shared__ float s[128];
  const long row = blockIdx.x;
  const int d = threadIdx.x;
  float v = in[row * D_ + d];
  s[d] = v; __syncthreads();
  for (int o = 64; o > 0; o >>= 1) { if (d < o) s[d] += s[d + o]; __syncthreads(); }
  float mu = s[0] * (1.f / 128.f);
  __syncthreads();
  float diff = v - mu;
  s[d] = diff * diff; __syncthreads();
  for (int o = 64; o > 0; o >>= 1) { if (d < o) s[d] += s[d + o]; __syncthreads(); }
  float var = s[0] * (1.f / 128.f);
  float y = diff * rsqrtf(var + 1e-5f) * g[d] + bta[d];
  xo[row * D_ + d]  = y;
  xbo[row * D_ + d] = (__bf16)y;
}

// f32 [K][N]  ->  bf16 [N][K]  (transpose + convert, done once for weights)
__global__ void f2bf_t_k(const float* __restrict__ s, __bf16* __restrict__ d,
                         int K, int N)
{
  int i = blockIdx.x * 256 + threadIdx.x;
  if (i < K * N) {
    int n = i / K, k = i - n * K;
    d[i] = (__bf16)s[(long)k * N + n];
  }
}

__global__ __launch_bounds__(256)
void head_k(const float* __restrict__ x, const float* __restrict__ Wm,
            const float* __restrict__ bm, float* __restrict__ out)
{
  __shared__ float s[256];
  const int blk = blockIdx.x;
  const int b = blk / 6, o = blk % 6;
  float acc = 0.f;
  for (int j = threadIdx.x; j < S_ * D_; j += 256)
    acc += x[(long)b * S_ * D_ + j] * Wm[(long)j * 6 + o];
  s[threadIdx.x] = acc; __syncthreads();
  for (int off = 128; off > 0; off >>= 1) {
    if (threadIdx.x < off) s[threadIdx.x] += s[threadIdx.x + off];
    __syncthreads();
  }
  if (threadIdx.x == 0) out[b * 6 + o] = s[0] + bm[o];
}

// ---------------------------------------------------------------------------
extern "C" void kernel_launch(void* const* d_in, const int* in_sizes, int n_in,
                              void* d_out, int out_size, void* d_ws, size_t ws_size,
                              hipStream_t stream)
{
  const int*   tok = (const int*)  d_in[0];
  const float* emb = (const float*)d_in[1];
  const float* pe  = (const float*)d_in[2];
  const float* hyp = (const float*)d_in[3];
  const float* Wq  = (const float*)d_in[4];  const float* bq = (const float*)d_in[5];
  const float* Wv  = (const float*)d_in[6];  const float* bv = (const float*)d_in[7];
  const float* lng = (const float*)d_in[8];  const float* lnb = (const float*)d_in[9];
  const float* W1  = (const float*)d_in[10]; const float* b1 = (const float*)d_in[11];
  const float* W2  = (const float*)d_in[12]; const float* b2 = (const float*)d_in[13];
  const float* Wm  = (const float*)d_in[14]; const float* bm = (const float*)d_in[15];
  float* out = (float*)d_out;

  // workspace carve-up (~24.5 MB total)
  char* W = (char*)d_ws;
  size_t off = 0;
  auto carve = [&](size_t bytes) { void* p = W + off; off += (bytes + 255) & ~(size_t)255; return p; };
  float*  x    = (float*) carve((size_t)B_*S_*D_*4);      // 2 MB
  __bf16* xb   = (__bf16*)carve((size_t)B_*S_*D_*2);      // 1 MB
  float*  q    = (float*) carve((size_t)B_*S_*D_*4);      // 2 MB
  __bf16* qb   = (__bf16*)carve((size_t)B_*S_*D_*2);      // 1 MB
  __bf16* vb   = (__bf16*)carve((size_t)B_*S_*D_*2);      // 1 MB
  int*    bkts = (int*)   carve((size_t)B_*S_*4);         // 16 KB
  float*  Lg   = (float*) carve((size_t)B_*S_*S_*4);      // 8 MB
  float*  attn = (float*) carve((size_t)B_*S_*D_*4);      // 2 MB
  __bf16* hb   = (__bf16*)carve((size_t)B_*S_*HID_*2);    // 4 MB
  float*  ffo  = (float*) carve((size_t)B_*S_*D_*4);      // 2 MB
  __bf16* Wqb  = (__bf16*)carve((size_t)D_*D_*2);         // transposed [N][K]
  __bf16* Wvb  = (__bf16*)carve((size_t)D_*D_*2);
  __bf16* W1b  = (__bf16*)carve((size_t)D_*HID_*2);       // [512][128]
  __bf16* W2b  = (__bf16*)carve((size_t)HID_*D_*2);       // [128][512]

  const long sTok = (long)S_ * D_;
  const float scale = 0.08838834764831845f;   // 1/sqrt(128)

  // weight downconvert + transpose to [N][K]
  f2bf_t_k<<<(D_*D_ + 255)/256,   256, 0, stream>>>(Wq, Wqb, D_, D_);
  f2bf_t_k<<<(D_*D_ + 255)/256,   256, 0, stream>>>(Wv, Wvb, D_, D_);
  f2bf_t_k<<<(D_*HID_ + 255)/256, 256, 0, stream>>>(W1, W1b, D_, HID_);
  f2bf_t_k<<<(HID_*D_ + 255)/256, 256, 0, stream>>>(W2, W2b, HID_, D_);

  embed_k<<<B_*S_, 128, 0, stream>>>(tok, emb, pe, x, xb);

  for (int enc = 0; enc < 2; ++enc) {
    gemm_bf16<<<dim3(2, 64, 1), 128, 0, stream>>>(xb, Wqb, bq, nullptr, q, qb,
        B_*S_, D_, D_, D_, D_, D_, 0, 0, 0, 0, 1.0f);
    gemm_bf16<<<dim3(2, 64, 1), 128, 0, stream>>>(xb, Wvb, bv, nullptr, nullptr, vb,
        B_*S_, D_, D_, D_, D_, D_, 0, 0, 0, 0, 1.0f);

    bucket_k<<<B_*S_/128, 128, 0, stream>>>(q, hyp, bkts);

    // bucket-independent logits: Lg[b] = scale * q[b] @ q[b]^T  (qb is [N][K])
    gemm_bf16<<<dim3(8, 8, B_), 128, 0, stream>>>(qb, qb, nullptr, nullptr, Lg, nullptr,
        S_, S_, D_, D_, D_, S_, sTok, sTok, (long)S_*S_, 0, scale);

    // 32-bucket masked softmax + P@V (185 KB dynamic LDS, TDM-fed logits)
    lsh_attn<<<dim3(S_/16, B_), 128, LDS_TOT, stream>>>(Lg, vb, bkts, attn);

    layernorm_k<<<B_*S_, 128, 0, stream>>>(attn, lng, lnb, x, xb);

    gemm_bf16<<<dim3(8, 64, 1), 128, 0, stream>>>(xb, W1b, b1, nullptr, nullptr, hb,
        B_*S_, HID_, D_, D_, D_, HID_, 0, 0, 0, 1, 1.0f);
    gemm_bf16<<<dim3(2, 64, 1), 128, 0, stream>>>(hb, W2b, b2, x, ffo, nullptr,
        B_*S_, D_, HID_, HID_, HID_, D_, 0, 0, 0, 0, 1.0f);

    layernorm_k<<<B_*S_, 128, 0, stream>>>(ffo, lng, lnb, x, xb);
  }

  head_k<<<B_*6, 256, 0, stream>>>(x, Wm, bm, out);
}
